// SpikingCollisionNavigationMultiHead_27616639713878
// MI455X (gfx1250) — matrix operand order repaired and checked by reading, not compiled, not run on hardware
//
#include <hip/hip_runtime.h>

// Problem dims (fixed by setup_inputs)
#define T_STEPS 256
#define BATCH   128
#define DIN     512
#define HID     2048
#define CHD     512
#define NHD     512
#define HHD     1024   // CHD + NHD (fused head width)
#define CD      2
#define ND      4

#define BETA   0.9f
#define THRESH 1.0f

typedef __attribute__((ext_vector_type(16))) __bf16 v16bf;
typedef __attribute__((ext_vector_type(8)))  __bf16 v8bf;
typedef __attribute__((ext_vector_type(8)))  float  v8f;
typedef __attribute__((ext_vector_type(4)))  unsigned v4u;
typedef __attribute__((ext_vector_type(8)))  int      v8i;
typedef __attribute__((ext_vector_type(4)))  int      v4i;

__device__ __forceinline__ __bf16 f2bf(float f) {
  union { float f; unsigned u; } v; v.f = f;
  unsigned r = v.u + 0x7FFFu + ((v.u >> 16) & 1u);   // round-to-nearest-even
  unsigned short h = (unsigned short)(r >> 16);
  return __builtin_bit_cast(__bf16, h);
}
__device__ __forceinline__ float bf2f(__bf16 b) {
  unsigned short h = __builtin_bit_cast(unsigned short, b);
  union { unsigned u; float f; } v; v.u = ((unsigned)h) << 16;
  return v.f;
}

// ---- TDM: DMA a [rows x K] bf16 tile (row stride K elems) from global into LDS ----
__device__ __forceinline__ void tdm_load_A(const __bf16* gbase, unsigned ldsAddr,
                                           int K, int rows) {
  unsigned long long ga = (unsigned long long)(size_t)gbase;
  v4u g0;
  g0[0] = 1u;                                              // count=1, user mode
  g0[1] = ldsAddr;                                         // lds_addr (bytes)
  g0[2] = (unsigned)(ga & 0xFFFFFFFFull);                  // global_addr[31:0]
  g0[3] = (unsigned)((ga >> 32) & 0x1FFFFFFull) | (2u << 30); // addr[56:32] | type=2
  v8i g1;
  g1[0] = (int)(1u << 16);                                 // data_size = 2 bytes
  g1[1] = (int)(((unsigned)K & 0xFFFFu) << 16);            // tensor_dim0 lo16
  g1[2] = (int)((((unsigned)K >> 16) & 0xFFFFu) |
                (((unsigned)rows & 0xFFFFu) << 16));       // dim0 hi16 | tensor_dim1
  g1[3] = (int)(((unsigned)K & 0xFFFFu) << 16);            // dim1 hi16=0 | tile_dim0=K
  g1[4] = rows;                                            // tile_dim1=rows, tile_dim2=0
  g1[5] = K;                                               // tensor_dim0_stride lo32
  g1[6] = 0;                                               // stride0 hi16 | stride1 lo16
  g1[7] = 0;                                               // stride1 hi32
  v4i z = {};
#if defined(__clang_major__) && (__clang_major__ >= 23)
  v8i z8 = {};
  __builtin_amdgcn_tensor_load_to_lds(g0, g1, z, z, z8, 0);
#else
  __builtin_amdgcn_tensor_load_to_lds(g0, g1, z, z, 0);
#endif
}

// ---------------- init / conversion kernels ----------------

__global__ void k_zero(float* __restrict__ p, int n) {
  int i = blockIdx.x * blockDim.x + threadIdx.x;
  for (; i < n; i += gridDim.x * blockDim.x) p[i] = 0.0f;
}

__global__ void k_cvt_x(const float* __restrict__ x, __bf16* __restrict__ xb, int n) {
  int i = blockIdx.x * blockDim.x + threadIdx.x;
  for (; i < n; i += gridDim.x * blockDim.x) xb[i] = f2bf(x[i]);
}

// WsT[h][k] = bf16(Ws[k][h]);  Ws is [DIN, HID]
__global__ void k_cvt_ws(const float* __restrict__ Ws, __bf16* __restrict__ WsT) {
  int i = blockIdx.x * blockDim.x + threadIdx.x;
  int n = HID * DIN;
  for (; i < n; i += gridDim.x * blockDim.x) {
    int h = i / DIN, k = i - h * DIN;
    WsT[i] = f2bf(Ws[(size_t)k * HID + h]);
  }
}

// WheadT[n][k]: n<CHD -> Wc[k][n], else Wn[k][n-CHD];  Wc/Wn are [HID, 512]
__global__ void k_cvt_whead(const float* __restrict__ Wc, const float* __restrict__ Wn,
                            __bf16* __restrict__ WheadT) {
  int i = blockIdx.x * blockDim.x + threadIdx.x;
  int tot = HHD * HID;
  for (; i < tot; i += gridDim.x * blockDim.x) {
    int n = i / HID, k = i - n * HID;
    float v = (n < CHD) ? Wc[(size_t)k * CHD + n] : Wn[(size_t)k * NHD + (n - CHD)];
    WheadT[i] = f2bf(v);
  }
}

__global__ void k_cvt_bhead(const float* __restrict__ bc, const float* __restrict__ bn,
                            float* __restrict__ bhead) {
  int i = blockIdx.x * blockDim.x + threadIdx.x;
  if (i < HHD) bhead[i] = (i < CHD) ? bc[i] : bn[i - CHD];
}

// ------------- fused GEMM + LIF:  I = A @ WT' + bias ; LIF(V) -> spikes -------------
// block = 64 threads (2 waves). Block computes 32(M) x 128(N); wave w owns 64 cols,
// both M-tiles. A tile [32 x K] staged in LDS via one TDM descriptor, shared by both
// waves. Each B fragment feeds 2 WMMAs (M-reuse), each A fragment feeds 4 (N-reuse).

template<int K>
__global__ __launch_bounds__(64) void k_gemm_lif(
    const __bf16* __restrict__ A,    // [BATCH, K]  bf16
    const __bf16* __restrict__ WT,   // [N, K]      bf16 (transposed weights)
    const float* __restrict__ bias,  // [N]
    float* __restrict__ V,           // [BATCH, N]
    __bf16* __restrict__ sout,       // [BATCH, N]
    int N, float* __restrict__ ssum, int sumBase, int sumThresh)
{
  extern __shared__ __bf16 smem[];   // [32][K]
  const int tid  = threadIdx.x;
  const int lane = tid & 31;
  const int wave = tid >> 5;
  const int half = lane >> 4;
  const int l16  = lane & 15;
  const int m0 = blockIdx.y * 32;
  const int n0 = blockIdx.x * 128 + wave * 64;

  if (tid < 32) {  // wave 0 drives the tensor DMA for the shared A tile
    tdm_load_A(A + (size_t)m0 * K, (unsigned)(size_t)(void*)smem, K, 32);
    __builtin_amdgcn_s_wait_tensorcnt(0);
  }
  __syncthreads();

  // Hoisted base pointers: K-loop indexes with immediate offsets after unrolling.
  const __bf16* arow0 = smem + (size_t)l16 * K + half * 16;         // rows m0..m0+15
  const __bf16* arow1 = smem + (size_t)(16 + l16) * K + half * 16;  // rows m0+16..m0+31
  const __bf16* wb0 = WT + (size_t)(n0 +  0 + l16) * K + half * 16;
  const __bf16* wb1 = WT + (size_t)(n0 + 16 + l16) * K + half * 16;
  const __bf16* wb2 = WT + (size_t)(n0 + 32 + l16) * K + half * 16;
  const __bf16* wb3 = WT + (size_t)(n0 + 48 + l16) * K + half * 16;

  v8f acc[2][4] = {};

#pragma unroll 4
  for (int k0 = 0; k0 < K; k0 += 32) {
    // A fragments from LDS: lane=row, 16 consecutive K per half-wave (half*16 .. half*16+15)
    v8bf lo0 = *(const v8bf*)(arow0 + k0);
    v8bf hi0 = *(const v8bf*)(arow0 + k0 + 8);
    v8bf lo1 = *(const v8bf*)(arow1 + k0);
    v8bf hi1 = *(const v8bf*)(arow1 + k0 + 8);
    v16bf a0 = __builtin_shufflevector(lo0, hi0, 0, 1, 2, 3, 4, 5, 6, 7,
                                       8, 9, 10, 11, 12, 13, 14, 15);
    v16bf a1 = __builtin_shufflevector(lo1, hi1, 0, 1, 2, 3, 4, 5, 6, 7,
                                       8, 9, 10, 11, 12, 13, 14, 15);
    // B fragments: lane=col, 16 consecutive K per half-wave; each feeds 2 WMMAs
    v16bf b0 = *(const v16bf*)(wb0 + k0);
    acc[0][0] = __builtin_amdgcn_wmma_f32_16x16x32_bf16(false, a0, false, b0, (short)0, acc[0][0], false, false);
    acc[1][0] = __builtin_amdgcn_wmma_f32_16x16x32_bf16(false, a1, false, b0, (short)0, acc[1][0], false, false);
    v16bf b1 = *(const v16bf*)(wb1 + k0);
    acc[0][1] = __builtin_amdgcn_wmma_f32_16x16x32_bf16(false, a0, false, b1, (short)0, acc[0][1], false, false);
    acc[1][1] = __builtin_amdgcn_wmma_f32_16x16x32_bf16(false, a1, false, b1, (short)0, acc[1][1], false, false);
    v16bf b2 = *(const v16bf*)(wb2 + k0);
    acc[0][2] = __builtin_amdgcn_wmma_f32_16x16x32_bf16(false, a0, false, b2, (short)0, acc[0][2], false, false);
    acc[1][2] = __builtin_amdgcn_wmma_f32_16x16x32_bf16(false, a1, false, b2, (short)0, acc[1][2], false, false);
    v16bf b3 = *(const v16bf*)(wb3 + k0);
    acc[0][3] = __builtin_amdgcn_wmma_f32_16x16x32_bf16(false, a0, false, b3, (short)0, acc[0][3], false, false);
    acc[1][3] = __builtin_amdgcn_wmma_f32_16x16x32_bf16(false, a1, false, b3, (short)0, acc[1][3], false, false);
  }

  // Epilogue: bias + LIF + spike emit.  C layout: lane=N, VGPR r -> M = r + 8*half.
  float lsum = 0.0f;
#pragma unroll
  for (int mt = 0; mt < 2; ++mt) {
#pragma unroll
    for (int s = 0; s < 4; ++s) {
      const int col = n0 + s * 16 + l16;
      const float bb = bias[col];
#pragma unroll
      for (int r = 0; r < 8; ++r) {
        const int row = m0 + mt * 16 + r + 8 * half;
        const size_t idx = (size_t)row * N + col;
        float v = V[idx];
        v = BETA * v + acc[mt][s][r] + bb;
        const float sp = (v - THRESH) > 0.0f ? 1.0f : 0.0f;
        V[idx] = v - sp * THRESH;
        sout[idx] = f2bf(sp);
        lsum += sp;
      }
    }
  }
  // wave-level reduction, single atomic per wave
#pragma unroll
  for (int off = 16; off > 0; off >>= 1) lsum += __shfl_down(lsum, off, 32);
  if (lane == 0) {
    const int which = sumBase + ((blockIdx.x * 128 >= sumThresh) ? 1 : 0);
    __hip_atomic_fetch_add(&ssum[which], lsum, __ATOMIC_RELAXED, __HIP_MEMORY_SCOPE_AGENT);
  }
}

// ---------------- tiny readouts + sequence writes + running means ----------------
// d_out layout (floats): [0,256) coll mean | [256,768) nav mean |
// [768, 768+65536) coll_seq | [66304, 197376) nav_seq | [197376,197379) spike_rate

__global__ __launch_bounds__(256) void k_out(
    const __bf16* __restrict__ shead,
    const float* __restrict__ Wch, const float* __restrict__ bch,
    const float* __restrict__ Wnh, const float* __restrict__ bnh,
    float* __restrict__ out, int t)
{
  int o = blockIdx.x * blockDim.x + threadIdx.x;
  if (o >= BATCH * (CD + ND)) return;
  int b = o / (CD + ND), j = o - b * (CD + ND);
  const __bf16* srow = shead + (size_t)b * HHD;
  if (j < CD) {
    float acc = bch[j];
    for (int c = 0; c < CHD; ++c) acc += bf2f(srow[c]) * Wch[c * CD + j];
    out[768 + (size_t)t * (BATCH * CD) + b * CD + j] = acc;
    out[b * CD + j] += acc * (1.0f / T_STEPS);
  } else {
    int jj = j - CD;
    float acc = bnh[jj];
    for (int c = 0; c < NHD; ++c) acc += bf2f(srow[CHD + c]) * Wnh[c * ND + jj];
    out[768 + BATCH * CD * T_STEPS + (size_t)t * (BATCH * ND) + b * ND + jj] = acc;
    out[BATCH * CD + b * ND + jj] += acc * (1.0f / T_STEPS);
  }
}

__global__ void k_fin(const float* __restrict__ ssum, float* __restrict__ out) {
  int i = threadIdx.x;
  if (i < 3) {
    float denom = (i == 0) ? (float)T_STEPS * (BATCH * HID)
                           : (float)T_STEPS * (BATCH * CHD);   // CHD == NHD
    out[768 + BATCH * CD * T_STEPS + BATCH * ND * T_STEPS + i] = ssum[i] / denom;
  }
}

// ---------------- host ----------------

#define CDIV(a, b) (((a) + (b) - 1) / (b))

extern "C" void kernel_launch(void* const* d_in, const int* in_sizes, int n_in,
                              void* d_out, int out_size, void* d_ws, size_t ws_size,
                              hipStream_t stream) {
  const float* x_seq = (const float*)d_in[0];
  const float* Ws  = (const float*)d_in[1];
  const float* bs  = (const float*)d_in[2];
  const float* Wc  = (const float*)d_in[3];
  const float* bc  = (const float*)d_in[4];
  const float* Wn  = (const float*)d_in[5];
  const float* bn  = (const float*)d_in[6];
  const float* Wch = (const float*)d_in[7];
  const float* bch = (const float*)d_in[8];
  const float* Wnh = (const float*)d_in[9];
  const float* bnh = (const float*)d_in[10];
  float* out = (float*)d_out;

  // Workspace carve-up
  char* w = (char*)d_ws;
  __bf16* xb     = (__bf16*)w; w += (size_t)T_STEPS * BATCH * DIN * 2;  // 32 MB
  __bf16* WsT    = (__bf16*)w; w += (size_t)HID * DIN * 2;              // 2 MB
  __bf16* WheadT = (__bf16*)w; w += (size_t)HHD * HID * 2;              // 4 MB
  float*  bhead  = (float*)w;  w += (size_t)HHD * 4;
  float*  Vs     = (float*)w;  w += (size_t)BATCH * HID * 4;            // 1 MB
  float*  Vhead  = (float*)w;  w += (size_t)BATCH * HHD * 4;            // 512 KB
  __bf16* ssb    = (__bf16*)w; w += (size_t)BATCH * HID * 2;            // 512 KB
  __bf16* sheadb = (__bf16*)w; w += (size_t)BATCH * HHD * 2;            // 256 KB
  float*  ssum   = (float*)w;  w += 256;

  // Init state + output accumulators; convert x and weights to bf16 once
  k_zero<<<CDIV(BATCH * HID, 256), 256, 0, stream>>>(Vs, BATCH * HID);
  k_zero<<<CDIV(BATCH * HHD, 256), 256, 0, stream>>>(Vhead, BATCH * HHD);
  k_zero<<<1, 32, 0, stream>>>(ssum, 3);
  k_zero<<<CDIV(BATCH * (CD + ND), 256), 256, 0, stream>>>(out, BATCH * (CD + ND));
  k_cvt_x<<<4096, 256, 0, stream>>>(x_seq, xb, T_STEPS * BATCH * DIN);
  k_cvt_ws<<<CDIV(HID * DIN, 256), 256, 0, stream>>>(Ws, WsT);
  k_cvt_whead<<<CDIV(HHD * HID, 256), 256, 0, stream>>>(Wc, Wn, WheadT);
  k_cvt_bhead<<<CDIV(HHD, 256), 256, 0, stream>>>(bc, bn, bhead);

  for (int t = 0; t < T_STEPS; ++t) {
    const __bf16* x_t = xb + (size_t)t * BATCH * DIN;
    // trunk: [128,512] @ [512,2048] ; all spikes -> ssum[0]
    k_gemm_lif<DIN><<<dim3(HID / 128, BATCH / 32), 64, 32 * DIN * 2, stream>>>(
        x_t, WsT, bs, Vs, ssb, HID, ssum, 0, 1 << 30);
    // heads: [128,2048] @ [2048,1024] ; ssum[1] (coll, n<512) / ssum[2] (nav)
    k_gemm_lif<HID><<<dim3(HHD / 128, BATCH / 32), 64, 32 * HID * 2, stream>>>(
        ssb, WheadT, bhead, Vhead, sheadb, HHD, ssum, 1, CHD);
    k_out<<<CDIV(BATCH * (CD + ND), 256), 256, 0, stream>>>(
        sheadb, Wch, bch, Wnh, bnh, out, t);
  }
  k_fin<<<1, 32, 0, stream>>>(ssum, out);
}
